// GAT_43533788512516
// MI455X (gfx1250) — matrix operand order repaired and checked by reading, not compiled, not run on hardware
//
#include <hip/hip_runtime.h>
#include <math.h>

// ---------------------------------------------------------------------------
// GAT (2-layer) forward for MI455X / gfx1250, wave32, WMMA f16->f32.
// ---------------------------------------------------------------------------

typedef _Float16 v16h __attribute__((ext_vector_type(16)));
typedef _Float16 h8   __attribute__((ext_vector_type(8)));
typedef float    v8f  __attribute__((ext_vector_type(8)));

__device__ __forceinline__ void atomicMaxF(float* addr, float value) {
  // Standard monotonic-bits trick; memory initialized to -inf.
  if (value >= 0.0f)
    atomicMax((int*)addr, __float_as_int(value));
  else
    atomicMin((unsigned int*)addr, __float_as_uint(value));
}

// ---------------- f32 -> f16 convert ----------------
__global__ void GAT_cvt_f16(const float* __restrict__ src,
                            _Float16* __restrict__ dst, int n) {
  int tid = blockIdx.x * blockDim.x + threadIdx.x;
  if (tid < n) dst[tid] = (_Float16)src[tid];
}

// ---------------- W[K,Nin] -> Wt[Nt,K] (f16, zero-padded rows) ----------------
__global__ void GAT_wt(const float* __restrict__ W, _Float16* __restrict__ Wt,
                       int K, int Nin, int Nt) {
  int tid = blockIdx.x * blockDim.x + threadIdx.x;
  if (tid >= Nt * K) return;
  int n = tid / K, k = tid % K;
  Wt[(size_t)n * K + k] = (n < Nin) ? (_Float16)W[(size_t)k * Nin + n]
                                    : (_Float16)0.0f;
}

// ---------------- WMMA GEMM: C[M, NS] = A[M,K](f16) * Bt[>=NS,K]^T(f16) ------
// One wave per 16x16 tile. K and output stride NS are compile-time:
// fully unrolled v_wmma chain, unguarded clause-able stores (C padded to NS).
template <int K, int NS>
__global__ __launch_bounds__(32)
void GAT_gemm_wmma(const _Float16* __restrict__ A,
                   const _Float16* __restrict__ Bt,
                   float* __restrict__ C, int M) {
  const int lane = threadIdx.x;
  const int half = lane >> 4;   // 0 | 1
  const int l16  = lane & 15;
  const int m0 = blockIdx.x * 16;
  const int n0 = blockIdx.y * 16;

  int arow = m0 + l16; if (arow >= M) arow = M - 1;   // clamp (M%16==0 normally)
  const _Float16* Ap = A  + (size_t)arow       * K;
  const _Float16* Bp = Bt + (size_t)(n0 + l16) * K;   // Bt padded, no OOB

  v8f acc = {};
#pragma unroll
  for (int k0 = 0; k0 < K; k0 += 32) {
    // A fragment: lane holds K = {half*8 + 0..7} and {16 + half*8 + 0..7}
    h8 a_lo = *(const h8*)(Ap + k0 + half * 8);
    h8 a_hi = *(const h8*)(Ap + k0 + 16 + half * 8);
    // B fragment: lane holds K = {half*8 + 0..7} and {16 + half*8 + 0..7} of column n
    h8 b_lo = *(const h8*)(Bp + k0 + half * 8);
    h8 b_hi = *(const h8*)(Bp + k0 + 16 + half * 8);
    v16h a, b;
#pragma unroll
    for (int i = 0; i < 8; ++i) {
      a[i] = a_lo[i]; a[i + 8] = a_hi[i];
      b[i] = b_lo[i]; b[i + 8] = b_hi[i];
    }
    acc = __builtin_amdgcn_wmma_f32_16x16x32_f16(
        /*neg_a=*/false, a, /*neg_b=*/false, b,
        /*c_mod=*/(short)0, acc, /*reuse_a=*/false, /*reuse_b=*/false);
  }
  // D layout: vgpr r -> M = r + 8*half, N = l16. Column always in-bounds (NS pad).
  float* Crow = C + (size_t)(m0 + half * 8) * NS + (n0 + l16);
  if (m0 + 16 <= M) {           // uniform fast path: full tile
#pragma unroll
    for (int r = 0; r < 8; ++r) Crow[(size_t)r * NS] = acc[r];
  } else {                      // ragged M tail (unused for M%16==0)
#pragma unroll
    for (int r = 0; r < 8; ++r)
      if (m0 + half * 8 + r < M) Crow[(size_t)r * NS] = acc[r];
  }
}

// ---------------- attention coefficients: AS/AD[n,h] = <F[n,h,:], a[h,:]> ----
// F has padded row stride FS (>= Hd*C).
__global__ void GAT_alpha(const float* __restrict__ F,
                          const float* __restrict__ asrc,
                          const float* __restrict__ adst,
                          float* __restrict__ AS, float* __restrict__ AD,
                          int N, int Hd, int C, int FS) {
  int tid = blockIdx.x * blockDim.x + threadIdx.x;
  if (tid >= N * Hd) return;
  int n = tid / Hd, h = tid % Hd;
  const float* row = F + (size_t)n * FS + (size_t)h * C;
  const float* ws = asrc + h * C;
  const float* wd = adst + h * C;
  float ss = 0.f, sd = 0.f;
  for (int c = 0; c < C; ++c) { float v = row[c]; ss += v * ws[c]; sd += v * wd[c]; }
  AS[tid] = ss; AD[tid] = sd;
}

__global__ void GAT_init_softmax(float* __restrict__ MX, float* __restrict__ SM, int n) {
  int tid = blockIdx.x * blockDim.x + threadIdx.x;
  if (tid < n) { MX[tid] = -INFINITY; SM[tid] = 0.0f; }
}

__global__ void GAT_zero(float* __restrict__ p, int n) {
  int tid = blockIdx.x * blockDim.x + threadIdx.x;
  if (tid < n) p[tid] = 0.0f;
}

// ---------------- per-edge logits + segment max ----------------
__global__ void GAT_edge_logits(const int* __restrict__ src, const int* __restrict__ dst,
                                const float* __restrict__ AS, const float* __restrict__ AD,
                                float* __restrict__ EA, float* __restrict__ MX,
                                int E, int Hd) {
  int tid = blockIdx.x * blockDim.x + threadIdx.x;
  if (tid >= E * Hd) return;
  int e = tid / Hd, h = tid % Hd;
  int s = src[e], d = dst[e];
  float v = AS[(size_t)s * Hd + h] + AD[(size_t)d * Hd + h];
  v = (v > 0.0f) ? v : 0.2f * v;            // leaky_relu, slope 0.2
  EA[tid] = v;
  atomicMaxF(&MX[(size_t)d * Hd + h], v);
}

// ---------------- exp(logit - max) + segment sum ----------------
__global__ void GAT_edge_exp(const int* __restrict__ dst,
                             const float* __restrict__ MX,
                             float* __restrict__ EA, float* __restrict__ SM,
                             int E, int Hd) {
  int tid = blockIdx.x * blockDim.x + threadIdx.x;
  if (tid >= E * Hd) return;
  int e = tid / Hd, h = tid % Hd;
  int d = dst[e];
  float m = MX[(size_t)d * Hd + h];         // finite: this edge contributed
  float ex = expf(EA[tid] - m);
  EA[tid] = ex;
  atomicAdd(&SM[(size_t)d * Hd + h], ex);
}

// ---------------- weighted message scatter-add ----------------
// F row stride FS (padded); AGG row stride Hd*C (compact).
__global__ void GAT_edge_agg(const int* __restrict__ src, const int* __restrict__ dst,
                             const float* __restrict__ F, const float* __restrict__ EA,
                             const float* __restrict__ SM, float* __restrict__ AGG,
                             int E, int Hd, int C, int FS) {
  const int HdC = Hd * C;
  long long tid = (long long)blockIdx.x * blockDim.x + threadIdx.x;
  if (tid >= (long long)E * HdC) return;
  int e = (int)(tid / HdC);
  int j = (int)(tid % HdC);
  int h = j / C;
  int s = src[e], d = dst[e];
  float alpha = EA[(size_t)e * Hd + h] / (SM[(size_t)d * Hd + h] + 1e-16f);
  atomicAdd(&AGG[(size_t)d * HdC + j], F[(size_t)s * FS + j] * alpha);
}

// ---------------- layer1 finalize: bias (in place) + ELU -> f16 ----------------
__global__ void GAT_final1(float* __restrict__ emb1, const float* __restrict__ b1,
                           _Float16* __restrict__ Hh, int n, int Fdim) {
  int tid = blockIdx.x * blockDim.x + threadIdx.x;
  if (tid >= n) return;
  float v = emb1[tid] + b1[tid % Fdim];
  emb1[tid] = v;
  float e = (v > 0.0f) ? v : expm1f(v);     // elu, alpha=1
  Hh[tid] = (_Float16)e;
}

// ---------------- layer2 finalize: mean over heads + bias ----------------
__global__ void GAT_final2(const float* __restrict__ AGG2, const float* __restrict__ b2,
                           float* __restrict__ emb2, int N, int Hd, int C) {
  int tid = blockIdx.x * blockDim.x + threadIdx.x;
  if (tid >= N * C) return;
  int n = tid / C, c = tid % C;
  float s = 0.0f;
  for (int h = 0; h < Hd; ++h) s += AGG2[(size_t)n * Hd * C + (size_t)h * C + c];
  emb2[tid] = s / (float)Hd + b2[c];
}

// ---------------- masked cross-entropy ----------------
__global__ void GAT_loss(const float* __restrict__ emb2, const int* __restrict__ labels,
                         const int* __restrict__ mask, float* __restrict__ acc,
                         int N, int C) {
  int n = blockIdx.x * blockDim.x + threadIdx.x;
  if (n >= N) return;
  const float* row = emb2 + (size_t)n * C;
  float mx = row[0];
  for (int c = 1; c < C; ++c) mx = fmaxf(mx, row[c]);
  float s = 0.0f;
  for (int c = 0; c < C; ++c) s += expf(row[c] - mx);
  float lse = logf(s) + mx;
  float ce = lse - row[labels[n]];
  if (mask[n] != 0) {
    atomicAdd(&acc[0], ce);
    atomicAdd(&acc[1], 1.0f);
  }
}

__global__ void GAT_loss_write(const float* __restrict__ acc, float* __restrict__ out) {
  if (threadIdx.x == 0 && blockIdx.x == 0) out[0] = acc[0] / acc[1];
}

// ---------------------------------------------------------------------------
extern "C" void kernel_launch(void* const* d_in, const int* in_sizes, int n_in,
                              void* d_out, int out_size, void* d_ws, size_t ws_size,
                              hipStream_t stream) {
  const float* x   = (const float*)d_in[0];
  const int*   ei  = (const int*)d_in[1];
  const int*   lab = (const int*)d_in[2];
  const int*   msk = (const int*)d_in[3];
  const float* W1  = (const float*)d_in[4];
  const float* as1 = (const float*)d_in[5];
  const float* ad1 = (const float*)d_in[6];
  const float* b1  = (const float*)d_in[7];
  const float* W2  = (const float*)d_in[8];
  const float* as2 = (const float*)d_in[9];
  const float* ad2 = (const float*)d_in[10];
  const float* b2  = (const float*)d_in[11];

  const int IN = 128, H1 = 8, C1 = 16, H2 = 10, C2 = 10;
  const int F1D = H1 * C1;   // 128 (= stride, layer 1)
  const int F2D = H2 * C2;   // 100 (compact), padded stride 112 in F
  const int F2S = 112;
  const int N = in_sizes[0] / IN;   // 100000
  const int E = in_sizes[1] / 2;    // 800000
  const int* esrc = ei;
  const int* edst = ei + E;

  // ---- workspace carve (d_ws is 256B aligned) ----
  char* ws = (char*)d_ws;
  size_t off = 0;
  auto carve = [&](size_t bytes) -> char* {
    char* p = ws + off;
    off = (off + bytes + 255) & ~(size_t)255;
    return p;
  };
  _Float16* xh  = (_Float16*)carve((size_t)N * IN * 2);    // 25.6 MB
  _Float16* Hh  = (_Float16*)carve((size_t)N * IN * 2);    // 25.6 MB
  _Float16* W1t = (_Float16*)carve((size_t)128 * 128 * 2);
  _Float16* W2t = (_Float16*)carve((size_t)112 * 128 * 2); // padded 100->112
  float*    F   = (float*)carve((size_t)N * IN * 4);       // feat1 (128), feat2 (112 pad)
  float*    AS  = (float*)carve((size_t)N * 10 * 4);
  float*    AD  = (float*)carve((size_t)N * 10 * 4);
  float*    MX  = (float*)carve((size_t)N * 10 * 4);
  float*    SM  = (float*)carve((size_t)N * 10 * 4);
  float*    EA  = (float*)carve((size_t)E * 10 * 4);       // 32 MB
  float*    lossacc = (float*)carve(256);
  // AGG2 aliases xh|Hh (both dead after GEMM2): needs N*100*4 = 40MB <= 51.2MB
  float*    AGG2 = (float*)ws;

  float* out  = (float*)d_out;
  float* emb1 = out + 1;                       // [N,128]
  float* emb2 = out + 1 + (size_t)N * F1D;     // [N,10]

  const int B = 256;
  auto nb = [](long long n) { return (unsigned)((n + 255) / 256); };

  // -------- setup --------
  GAT_zero<<<1, 32, 0, stream>>>(lossacc, 2);
  GAT_cvt_f16<<<nb((long long)N * IN), B, 0, stream>>>(x, xh, N * IN);
  GAT_wt<<<nb(128 * 128), B, 0, stream>>>(W1, W1t, 128, 128, 128);
  GAT_wt<<<nb(112 * 128), B, 0, stream>>>(W2, W2t, 128, 100, 112);

  // -------- layer 1 --------
  dim3 g1((N + 15) / 16, 128 / 16);
  GAT_gemm_wmma<128, 128><<<g1, 32, 0, stream>>>(xh, W1t, F, N);
  GAT_alpha<<<nb((long long)N * H1), B, 0, stream>>>(F, as1, ad1, AS, AD, N, H1, C1, F1D);
  GAT_init_softmax<<<nb((long long)N * H1), B, 0, stream>>>(MX, SM, N * H1);
  GAT_edge_logits<<<nb((long long)E * H1), B, 0, stream>>>(esrc, edst, AS, AD, EA, MX, E, H1);
  GAT_edge_exp<<<nb((long long)E * H1), B, 0, stream>>>(edst, MX, EA, SM, E, H1);
  GAT_zero<<<nb((long long)N * F1D), B, 0, stream>>>(emb1, N * F1D);
  GAT_edge_agg<<<nb((long long)E * F1D), B, 0, stream>>>(esrc, edst, F, EA, SM, emb1,
                                                         E, H1, C1, F1D);
  GAT_final1<<<nb((long long)N * F1D), B, 0, stream>>>(emb1, b1, Hh, N * F1D, F1D);

  // -------- layer 2 (F padded to stride 112) --------
  dim3 g2((N + 15) / 16, 112 / 16);
  GAT_gemm_wmma<128, 112><<<g2, 32, 0, stream>>>(Hh, W2t, F, N);
  GAT_alpha<<<nb((long long)N * H2), B, 0, stream>>>(F, as2, ad2, AS, AD, N, H2, C2, F2S);
  GAT_init_softmax<<<nb((long long)N * H2), B, 0, stream>>>(MX, SM, N * H2);
  GAT_edge_logits<<<nb((long long)E * H2), B, 0, stream>>>(esrc, edst, AS, AD, EA, MX, E, H2);
  GAT_edge_exp<<<nb((long long)E * H2), B, 0, stream>>>(edst, MX, EA, SM, E, H2);
  GAT_zero<<<nb((long long)N * F2D), B, 0, stream>>>(AGG2, N * F2D);
  GAT_edge_agg<<<nb((long long)E * F2D), B, 0, stream>>>(esrc, edst, F, EA, SM, AGG2,
                                                         E, H2, C2, F2S);
  GAT_final2<<<nb((long long)N * C2), B, 0, stream>>>(AGG2, b2, emb2, N, H2, C2);

  // -------- loss --------
  GAT_loss<<<nb(N), B, 0, stream>>>(emb2, lab, msk, lossacc, N, C2);
  GAT_loss_write<<<1, 1, 0, stream>>>(lossacc, out);
}
